// SemanticIds_11785390260327
// MI455X (gfx1250) — matrix operand order repaired and checked by reading, not compiled, not run on hardware
//
#include <hip/hip_runtime.h>
#include <hip/hip_bf16.h>

// Problem dims (fixed by the reference)
#define NROWS_TOT 131072
#define D_IN 768
#define DD   512
#define KK   256
#define LL   4

#define ROWS 32          // rows per workgroup in the fused kernel
#define BLK  256         // 8 waves of 32

typedef __attribute__((ext_vector_type(16))) __bf16 v16bf;
typedef __attribute__((ext_vector_type(8)))  float  v8f;

union Frag { uint4 q[2]; v16bf v; };

// round-to-nearest-even f32 -> bf16 (pure VALU)
__device__ __forceinline__ unsigned short f2bf(float f) {
    unsigned u = __float_as_uint(f);
    u += 0x7FFFu + ((u >> 16) & 1u);
    return (unsigned short)(u >> 16);
}

// ---------------------------------------------------------------------------
// Precompute kernels (tiny: all operands L2-resident, ~0.1 GFLOP total)
// ---------------------------------------------------------------------------

// enc_w [768,512] f32 (K-major) -> Wt [512,768] bf16 (col-major): B fragments
// become 16B K-runs (two per lane) loadable straight from L2 with global b128s.
__global__ void transpose_bf16_kernel(const float* __restrict__ src,
                                      unsigned short* __restrict__ dst) {
    int i = blockIdx.x * blockDim.x + threadIdx.x;
    if (i < D_IN * DD) {
        int k = i / DD, n = i % DD;          // coalesced read, scattered write
        dst[n * D_IN + k] = f2bf(src[i]);
    }
}

__global__ void conv_bf16_kernel(const float* __restrict__ src,
                                 unsigned short* __restrict__ dst, int n) {
    int i = blockIdx.x * blockDim.x + threadIdx.x;
    if (i < n) dst[i] = f2bf(src[i]);
}

// norms[l*K+k] = ||codebooks[l][k]||^2
__global__ void norms_kernel(const float* __restrict__ cb,
                             float* __restrict__ norms) {
    int i = blockIdx.x * blockDim.x + threadIdx.x;
    if (i < LL * KK) {
        const float* row = cb + (size_t)i * DD;
        float s = 0.f;
        for (int d = 0; d < DD; ++d) s += row[d] * row[d];
        norms[i] = s;
    }
}

// T[(l-1)*K + kp] = argmin_k ( -2 * C_{l-1}[kp] . C_l[k] + ||C_l[k]||^2 )
__global__ void chain_kernel(const float* __restrict__ cb,
                             const float* __restrict__ norms,
                             int* __restrict__ T) {
    __shared__ float sv[KK];
    __shared__ int   si[KK];
    int b = blockIdx.x;
    int l = 1 + b / KK;
    int kp = b % KK;
    int k = threadIdx.x;
    const float* prev = cb + ((size_t)(l - 1) * KK + kp) * DD;
    const float* cur  = cb + ((size_t)l * KK + k) * DD;
    float dot = 0.f;
    for (int d = 0; d < DD; ++d) dot += prev[d] * cur[d];
    sv[k] = norms[l * KK + k] - 2.f * dot;
    si[k] = k;
    __syncthreads();
    for (int s = KK / 2; s > 0; s >>= 1) {
        if (k < s) {
            float a = sv[k], c = sv[k + s];
            if (c < a || (c == a && si[k + s] < si[k])) { sv[k] = c; si[k] = si[k + s]; }
        }
        __syncthreads();
    }
    if (k == 0) T[(l - 1) * KK + kp] = si[0];
}

// R[k] = C0[k] + C1[T1[k]] + C2[T2[T1[k]]] + C3[T3[T2[T1[k]]]]
__global__ void buildR_kernel(const float* __restrict__ cb,
                              const int* __restrict__ T,
                              float* __restrict__ R) {
    int k = blockIdx.x;
    int i1 = T[k];
    int i2 = T[KK + i1];
    int i3 = T[2 * KK + i2];
    const float* c0 = cb + (size_t)k * DD;
    const float* c1 = cb + ((size_t)KK * 1 + i1) * DD;
    const float* c2 = cb + ((size_t)KK * 2 + i2) * DD;
    const float* c3 = cb + ((size_t)KK * 3 + i3) * DD;
    for (int d = threadIdx.x; d < DD; d += blockDim.x)
        R[(size_t)k * DD + d] = c0[d] + c1[d] + c2[d] + c3[d];
}

// RW[k][j] = dec_b[j] + sum_d R[k][d] * dec_w[d][j]     (256x512x512, trivial)
__global__ void rw_kernel(const float* __restrict__ R,
                          const float* __restrict__ dec_w,
                          const float* __restrict__ dec_b,
                          float* __restrict__ RW) {
    int k = blockIdx.x;
    const float* r = R + (size_t)k * DD;
    for (int j = threadIdx.x; j < DD; j += blockDim.x) {
        float acc = dec_b[j];
        for (int d = 0; d < DD; ++d) acc += r[d] * dec_w[(size_t)d * DD + j];
        RW[(size_t)k * DD + j] = acc;
    }
}

// ---------------------------------------------------------------------------
// Fused main kernel: encoder GEMM (bf16 WMMA, B straight from L2-resident Wt)
// -> score GEMM vs C0 (bf16 WMMA) -> per-row argmin -> gather RW row.
// One read of X, one write of out.
// ---------------------------------------------------------------------------
__global__ __launch_bounds__(BLK) void fused_kernel(
    const float* __restrict__ X,               // [N, 768] f32
    const unsigned short* __restrict__ Wt,     // enc_w^T bf16 [512, 768]
    const float* __restrict__ enc_b,           // [512]
    const unsigned short* __restrict__ C0bf,   // codebook0 bf16 [256, 512]
    const float* __restrict__ norm0,           // [256]
    const float* __restrict__ RW,              // [256, 512] f32
    float* __restrict__ out)                   // [N, 512] f32
{
    __shared__ unsigned short Xs[ROWS * 128];  //  8 KB: X super-tile, bf16
    __shared__ unsigned short R0s[ROWS * DD];  // 32 KB: r0 block, bf16 row-major
    __shared__ float argval[ROWS * 4];
    __shared__ int   argidx[ROWS * 4];
    __shared__ int   idxfin[ROWS];

    const int tid  = threadIdx.x;
    const int lane = tid & 31;
    const int w    = tid >> 5;       // wave 0..7
    const int rs   = w >> 2;         // row stripe 0..1 (16 rows each)
    const int cq   = w & 3;          // col/code quarter
    const int l15  = lane & 15;
    const int hi   = lane >> 4;      // 0/1 lane half
    const size_t row0 = (size_t)blockIdx.x * ROWS;

    // staging coordinates (32 rows x 128 dims per super-tile, 16 f32/thread)
    const int sr  = tid >> 3;        // 0..31: row
    const int sd  = (tid & 7) * 16;  // 0..112: dim chunk base

    // ---------- Stage 1: r0 = X @ enc_w + enc_b (bf16 WMMA, f32 accum) -----
    v8f acc[8];
    {
        v8f z = {0.f,0.f,0.f,0.f,0.f,0.f,0.f,0.f};
        for (int ct = 0; ct < 8; ++ct) acc[ct] = z;
    }
    for (int st = 0; st < D_IN / 128; ++st) {          // 6 super-tiles
        // prefetch next X super-tile (global_prefetch_b8)
        if (st < D_IN / 128 - 1)
            __builtin_prefetch(X + (row0 + sr) * D_IN + (st + 1) * 128 + sd, 0, 1);
        // stage A super-tile: [32 rows x 128 dims] f32 -> bf16
        for (int i = 0; i < 2; ++i) {
            const float* p = X + (row0 + sr) * D_IN + st * 128 + sd + i * 8;
            float4 a = *(const float4*)(p);
            float4 b = *(const float4*)(p + 4);
            uint4 pk;
            pk.x = (unsigned)f2bf(a.x) | ((unsigned)f2bf(a.y) << 16);
            pk.y = (unsigned)f2bf(a.z) | ((unsigned)f2bf(a.w) << 16);
            pk.z = (unsigned)f2bf(b.x) | ((unsigned)f2bf(b.y) << 16);
            pk.w = (unsigned)f2bf(b.z) | ((unsigned)f2bf(b.w) << 16);
            *(uint4*)&Xs[sr * 128 + sd + i * 8] = pk;
        }
        __syncthreads();
        for (int ktl = 0; ktl < 4; ++ktl) {            // 32 dims each
            // A fragment (ISA 16-bit A 16x32 layout: two 16B runs per lane)
            Frag af;
            {
                int m = rs * 16 + l15;
                const char* ab = (const char*)Xs + m * 256 + ktl * 64 + hi * 16;
                af.q[0] = *(const uint4*)(ab);
                af.q[1] = *(const uint4*)(ab + 32);
            }
            const int kg = st * 128 + ktl * 32;        // global dim base
            for (int ct = 0; ct < 8; ++ct) {
                int n = cq * 128 + ct * 16 + l15;      // output column
                Frag bfr;                              // two 16B K-runs in Wt row n
                const char* bb = (const char*)Wt + (size_t)n * (D_IN * 2) + kg * 2 + hi * 16;
                bfr.q[0] = *(const uint4*)(bb);
                bfr.q[1] = *(const uint4*)(bb + 32);
                acc[ct] = __builtin_amdgcn_wmma_f32_16x16x32_bf16(
                    false, af.v, false, bfr.v, (short)0, acc[ct], false, false);
            }
        }
        __syncthreads();   // before next super-tile overwrites Xs
    }
    // epilogue: + enc_b, convert to bf16, store r0 block
    for (int ct = 0; ct < 8; ++ct) {
        int n = cq * 128 + ct * 16 + l15;
        float b = enc_b[n];
        for (int j = 0; j < 8; ++j) {
            int m = rs * 16 + j + hi * 8;
            R0s[m * DD + n] = f2bf(acc[ct][j] + b);
        }
    }
    __syncthreads();

    // ---------- Stage 2: scores = -2 * r0 @ C0^T + ||C0||^2, argmin --------
    v8f acc2[4];
    {
        v8f z = {0.f,0.f,0.f,0.f,0.f,0.f,0.f,0.f};
        for (int ct = 0; ct < 4; ++ct) acc2[ct] = z;
    }
    for (int kt = 0; kt < DD / 32; ++kt) {
        Frag af;
        {
            int m = rs * 16 + l15;
            const char* ab = (const char*)R0s + m * (DD * 2) + kt * 64 + hi * 16;
            af.q[0] = *(const uint4*)(ab);
            af.q[1] = *(const uint4*)(ab + 32);
        }
        for (int ct = 0; ct < 4; ++ct) {
            int n = cq * 64 + ct * 16 + l15;  // code index
            Frag bfr;                         // B(k,n) = C0[n][k]: contiguous in row n
            const char* bb = (const char*)C0bf + (size_t)n * (DD * 2) + kt * 64 + hi * 16;
            bfr.q[0] = *(const uint4*)(bb);
            bfr.q[1] = *(const uint4*)(bb + 32);
            acc2[ct] = __builtin_amdgcn_wmma_f32_16x16x32_bf16(
                false, af.v, false, bfr.v, (short)0, acc2[ct], false, false);
        }
    }
    // per-row argmin: rows live in lane halves; codes across lanes & ct tiles
    float mv[8]; int mi[8];
    for (int j = 0; j < 8; ++j) { mv[j] = 3.4e38f; mi[j] = 0x7fffffff; }
    for (int ct = 0; ct < 4; ++ct) {
        int n = cq * 64 + ct * 16 + l15;
        float nb = norm0[n];
        for (int j = 0; j < 8; ++j) {
            float s = -2.f * acc2[ct][j] + nb;
            if (s < mv[j] || (s == mv[j] && n < mi[j])) { mv[j] = s; mi[j] = n; }
        }
    }
    for (int d = 1; d < 16; d <<= 1) {       // xor-reduce within each 16-lane half
        for (int j = 0; j < 8; ++j) {
            float ov = __shfl_xor(mv[j], d, 32);
            int   oi = __shfl_xor(mi[j], d, 32);
            if (ov < mv[j] || (ov == mv[j] && oi < mi[j])) { mv[j] = ov; mi[j] = oi; }
        }
    }
    if (l15 == 0) {
        for (int j = 0; j < 8; ++j) {
            int m = rs * 16 + j + hi * 8;
            argval[m * 4 + cq] = mv[j];
            argidx[m * 4 + cq] = mi[j];
        }
    }
    __syncthreads();
    if (tid < ROWS) {
        float bv = argval[tid * 4]; int bi = argidx[tid * 4];
        for (int q = 1; q < 4; ++q) {
            float v = argval[tid * 4 + q]; int i = argidx[tid * 4 + q];
            if (v < bv || (v == bv && i < bi)) { bv = v; bi = i; }
        }
        idxfin[tid] = bi;
    }
    __syncthreads();

    // ---------- Stage 3: out[row] = RW[idx[row]] (dec bias folded in) ------
    {
        int r  = tid >> 3;            // 8 threads per row
        int c0 = (tid & 7) * 16;      // 16 float4 chunks each (128 per row)
        const float4* src = (const float4*)(RW + (size_t)idxfin[r] * DD) + c0;
        float4* dst = (float4*)(out + (row0 + r) * DD) + c0;
        for (int i = 0; i < 16; ++i) dst[i] = src[i];
    }
}

// ---------------------------------------------------------------------------
extern "C" void kernel_launch(void* const* d_in, const int* in_sizes, int n_in,
                              void* d_out, int out_size, void* d_ws, size_t ws_size,
                              hipStream_t stream) {
    const float* X     = (const float*)d_in[0];   // [N, 768]
    const float* enc_w = (const float*)d_in[1];   // [768, 512]
    const float* enc_b = (const float*)d_in[2];   // [512]
    const float* cb    = (const float*)d_in[3];   // [4, 256, 512]
    const float* dec_w = (const float*)d_in[4];   // [512, 512]
    const float* dec_b = (const float*)d_in[5];   // [512]
    float* out = (float*)d_out;

    // workspace layout (all offsets 256B-aligned; total ~2.1 MB)
    char* ws = (char*)d_ws;
    unsigned short* Wt   = (unsigned short*)(ws + 0);        // 786432 B  [512][768] bf16
    unsigned short* C0bf = (unsigned short*)(ws + 786432);   // 262144 B
    float* norms         = (float*)(ws + 1048576);           //   4096 B
    int*   T             = (int*)(ws + 1052672);             //   4096 B
    float* R             = (float*)(ws + 1056768);           // 524288 B
    float* RW            = (float*)(ws + 1581056);           // 524288 B

    // precompute (stream-ordered dependencies)
    transpose_bf16_kernel<<<(D_IN * DD + 255) / 256, 256, 0, stream>>>(enc_w, Wt);
    conv_bf16_kernel<<<(KK * DD + 255) / 256, 256, 0, stream>>>(cb, C0bf, KK * DD);
    norms_kernel<<<(LL * KK + 255) / 256, 256, 0, stream>>>(cb, norms);
    chain_kernel<<<3 * KK, KK, 0, stream>>>(cb, norms, T);
    buildR_kernel<<<KK, 256, 0, stream>>>(cb, T, R);
    rw_kernel<<<KK, 256, 0, stream>>>(R, dec_w, dec_b, RW);

    // fused main pass: one read of X, one write of out
    fused_kernel<<<NROWS_TOT / ROWS, BLK, 0, stream>>>(
        X, Wt, enc_b, C0bf, norms /* layer-0 norms are first 256 */, RW, out);
}
